// ContextualAttentionMask_45913200394447
// MI455X (gfx1250) — matrix operand order repaired and checked by reading, not compiled, not run on hardware
//
#include <hip/hip_runtime.h>
#include <hip/hip_bf16.h>

// ---------------------------------------------------------------------------
// ContextualAttentionMask on MI455X (gfx1250)
// Fused flash-attention formulation (padding in the reference is dead code):
//   Kn = rownorm(F^T + 1e-7)            (4096x256 per batch)
//   S  = Kn @ F                          -> softmax over keys j (axis 0)
//   O  = (Kn*m)^T @ softmax(S)           (256x4096 per batch)
//   out = O*(1-m) + F*m   (per-batch skip override)
// Two bf16 WMMA GEMM chains fused with online softmax; S never materialized.
// Key-tile operands (wave-invariant in the workgroup) are streamed into LDS
// by the Tensor Data Mover, double-buffered, overlapped with WMMA compute.
// Query-tile B operand is held resident in VGPRs across the whole key loop.
// ---------------------------------------------------------------------------

#define BATCH 4
#define CH    256
#define HW    4096          // 64*64

typedef __bf16 bf16;
typedef __attribute__((ext_vector_type(16))) __bf16 bf16x16;
typedef __attribute__((ext_vector_type(8)))  __bf16 bf16x8;
typedef __attribute__((ext_vector_type(8)))  float  f32x8;
typedef __attribute__((ext_vector_type(4)))  unsigned int u32x4;
typedef __attribute__((ext_vector_type(8)))  int    i32x8;
typedef __attribute__((ext_vector_type(4)))  int    i32x4;

__device__ __forceinline__ bf16x16 cat16(bf16x8 lo, bf16x8 hi) {
    return __builtin_shufflevector(lo, hi, 0,1,2,3,4,5,6,7,8,9,10,11,12,13,14,15);
}

// ---- Tensor Data Mover: 2D tile global->LDS with per-row LDS padding ------
// Descriptor per CDNA5 ISA (Group0 128b, Group1 256b).
// pad_interval_code: dwords-before-pad = 2<<code ; pad_amount_m1: dwords-1.
__device__ __forceinline__ void tdm_load_2d(const void* lds_dst,
                                            const void* global_src,
                                            unsigned tile_d0, unsigned tile_d1,
                                            unsigned tensor_d0, unsigned tensor_d1,
                                            unsigned stride0,
                                            unsigned pad_interval_code,
                                            unsigned pad_amount_m1) {
    const unsigned long long ga = (unsigned long long)(uintptr_t)global_src;
    u32x4 g0;
    g0[0] = 1u;                                           // count=1, user mode
    g0[1] = (unsigned)(uintptr_t)lds_dst;                 // lds_addr (bytes)
    g0[2] = (unsigned)ga;                                 // global_addr[31:0]
    g0[3] = (unsigned)((ga >> 32) & 0x01FFFFFFu)          // global_addr[56:32]
          | (2u << 30);                                   // type = 2 ("image")
    i32x8 g1;
    g1[0] = (int)((1u << 16)                              // data_size = 2 bytes
          |      (1u << 20)                               // pad_enable
          |      (pad_interval_code << 22)
          |      (pad_amount_m1 << 25));
    g1[1] = (int)((tensor_d0 & 0xFFFFu) << 16);           // abar=0 | dim0[15:0]
    g1[2] = (int)((tensor_d0 >> 16) | ((tensor_d1 & 0xFFFFu) << 16));
    g1[3] = (int)((tensor_d1 >> 16) | (tile_d0 << 16));   // dim1[31:16]|tile0
    g1[4] = (int)(tile_d1 & 0xFFFFu);                     // tile1 | tile2=0
    g1[5] = (int)stride0;                                 // dim0_stride[31:0]
    g1[6] = 0;                                            // stride0 hi | stride1 lo
    g1[7] = 0;
    const i32x4 z4 = {0, 0, 0, 0};
#if defined(__clang_major__) && (__clang_major__ >= 23)
    const i32x8 z8 = {0, 0, 0, 0, 0, 0, 0, 0};
    __builtin_amdgcn_tensor_load_to_lds(g0, g1, z4, z4, z8, 0);
#else
    __builtin_amdgcn_tensor_load_to_lds(g0, g1, z4, z4, 0);
#endif
}

// ---------------- prep 0: per-batch mask sum (for the skip predicate) ------
__global__ __launch_bounds__(256) void prep_masksum(const float* __restrict__ mask,
                                                    float* __restrict__ msum) {
    __shared__ float red[256];
    const int b = blockIdx.x;
    float s = 0.f;
    for (int i = threadIdx.x; i < HW; i += 256) s += mask[(size_t)b * HW + i];
    red[threadIdx.x] = s;
    __syncthreads();
    for (int st = 128; st > 0; st >>= 1) {
        if (threadIdx.x < st) red[threadIdx.x] += red[threadIdx.x + st];
        __syncthreads();
    }
    if (threadIdx.x == 0) msum[b] = red[0];
}

// ---------------- prep 1: inverse row norms of (F^T + eps) -----------------
__global__ __launch_bounds__(256) void prep_norm(const float* __restrict__ F,
                                                 const float* __restrict__ mask,
                                                 float* __restrict__ invn,
                                                 float* __restrict__ invnm) {
    const int b = blockIdx.y;
    const int j = blockIdx.x * 256 + threadIdx.x;
    const float* Fb = F + (size_t)b * CH * HW;
    float s = 0.f;
    #pragma unroll 8
    for (int c = 0; c < CH; ++c) {
        float v = Fb[(size_t)c * HW + j] + 1e-7f;   // coalesced along j
        s += v * v;
    }
    const float inv = rsqrtf(s);
    invn [(size_t)b * HW + j] = inv;
    invnm[(size_t)b * HW + j] = inv * mask[(size_t)b * HW + j];
}

// ---------------- prep 2: write bf16 operand tensors -----------------------
//   Kn_bf [b][j][c] = bf16((F[c][j]+eps) * invn[j])      (GEMM1 A operand)
//   FT_bf [b][i][c] = bf16(F[c][i])                      (GEMM1 B operand)
//   KmT_bf[b][c][j] = bf16((F[c][j]+eps) * invn[j]*m[j]) (GEMM2 A operand)
__global__ __launch_bounds__(256) void prep_transpose(const float* __restrict__ F,
                                                      const float* __restrict__ invn,
                                                      const float* __restrict__ invnm,
                                                      bf16* __restrict__ Kn,
                                                      bf16* __restrict__ FT,
                                                      bf16* __restrict__ KmT) {
    __shared__ float tile[32][33];
    const int b  = blockIdx.z;
    const int j0 = blockIdx.x * 32;
    const int c0 = blockIdx.y * 32;
    const int tx = threadIdx.x;      // 0..31 -> j in pass1, c in pass2
    const int ty = threadIdx.y;      // 0..7

    const float invnm_j = invnm[(size_t)b * HW + j0 + tx];
    #pragma unroll
    for (int r = 0; r < 4; ++r) {
        const int cl = ty + r * 8;
        const float v = F[((size_t)b * CH + c0 + cl) * HW + j0 + tx];
        tile[cl][tx] = v;
        KmT[((size_t)b * CH + c0 + cl) * HW + j0 + tx] = (bf16)((v + 1e-7f) * invnm_j);
    }
    __syncthreads();
    #pragma unroll
    for (int r = 0; r < 4; ++r) {
        const int jl = ty + r * 8;
        const int j  = j0 + jl;
        const float invn_j = invn[(size_t)b * HW + j];
        const float v = tile[tx][jl];
        const size_t o = ((size_t)b * HW + j) * CH + c0 + tx;
        Kn[o] = (bf16)((v + 1e-7f) * invn_j);
        FT[o] = (bf16)v;
    }
}

// ---------------- fused attention kernel -----------------------------------
// Block = 128 threads (4 waves); each wave owns a 16-column i-tile.
// TDM streams the key-step tiles (Kn 32x256, KmT 256x32) into double-buffered
// LDS with bank-spreading row padding; one wave issues DMA for step t+1 while
// all waves run 32 WMMAs on step t. F^T query operand resident in VGPRs.
//
// Dynamic LDS layout (bytes):
//   FT   : 64 rows * 544   = 34816   (256 elems + 16 pad elems per row)
//   Kn[2]: 32 rows * 528   = 16896*2 (256 elems +  8 pad elems per row)
//   KmT[2]: 256 rows * 80  = 20480*2 ( 32 elems +  8 pad elems per row)
#define FT_OFF   0
#define FT_SZ    (64 * 544)
#define KN_OFF   (FT_OFF + FT_SZ)
#define KN_SZ    (32 * 528)
#define KM_OFF   (KN_OFF + 2 * KN_SZ)
#define KM_SZ    (256 * 80)
#define SMEM_SZ  (KM_OFF + 2 * KM_SZ)

__global__ __launch_bounds__(128) void attn_kernel(const bf16*  __restrict__ Kn,
                                                   const bf16*  __restrict__ FT,
                                                   const bf16*  __restrict__ KmT,
                                                   const float* __restrict__ F,
                                                   const float* __restrict__ mask,
                                                   const float* __restrict__ msum,
                                                   float*       __restrict__ out) {
    extern __shared__ char smem[];

    const int b    = blockIdx.y;
    const int iblk = blockIdx.x * 64;
    const int tid  = threadIdx.x;
    const int wave = tid >> 5;
    const int lane = tid & 31;
    const int half = lane >> 4;       // which 16-lane half
    const int m16  = lane & 15;
    const int i0   = iblk + wave * 16;

    const bf16* knb = Kn  + (size_t)b * HW * CH;
    const bf16* kmb = KmT + (size_t)b * CH * HW;

    // ---- TDM prologue: query tile + first key tiles ----
    if (wave == 0) {
        // F^T query tile: 64 rows x 256, row pad 8 dwords -> 544B rows
        tdm_load_2d(smem + FT_OFF, FT + ((size_t)b * HW + iblk) * CH,
                    256, 64, 256, 64, 256, /*2<<6=128 dw*/6, /*8 dw*/7);
        // Kn key tile 0: 32 rows x 256, row pad 4 dwords -> 528B rows
        tdm_load_2d(smem + KN_OFF, knb, 256, 32, 256, 32, 256, 6, 3);
        // KmT key tile 0: 256 rows x 32 (stride HW), pad 4 dwords -> 80B rows
        tdm_load_2d(smem + KM_OFF, kmb, 32, 256, HW, 256, HW, /*2<<3=16 dw*/3, 3);
        __builtin_amdgcn_s_wait_tensorcnt(0);
    }
    __syncthreads();

    // ---- GEMM1 B operand: resident across the whole key loop (64 VGPRs) ----
    // Per lane-half: K=c {0..7,16..23}/{8..15,24..31} of query row i.
    const bf16* ldsFTrow = (const bf16*)(smem + FT_OFF)
                         + (wave * 16 + m16) * 272 + half * 8;
    bf16x16 bq[8];
    #pragma unroll
    for (int k = 0; k < 8; ++k)
        bq[k] = cat16(*(const bf16x8*)&ldsFTrow[k * 32],
                      *(const bf16x8*)&ldsFTrow[k * 32 + 16]);

    f32x8 acc[16];                    // O accumulator: 256 channels x 16 cols
    #pragma unroll
    for (int t = 0; t < 16; ++t) acc[t] = (f32x8){0.f,0.f,0.f,0.f,0.f,0.f,0.f,0.f};
    float m_run = -3.0e38f;
    float l_run = 0.f;

    for (int t = 0; t < HW / 32; ++t) {
        const int cur = t & 1;
        // ---- kick DMA for the next key tile (overlaps with compute) ----
        if (wave == 0 && (t + 1) < HW / 32) {
            const int jn = (t + 1) * 32;
            tdm_load_2d(smem + KN_OFF + (cur ^ 1) * KN_SZ,
                        knb + (size_t)jn * CH, 256, 32, 256, 32, 256, 6, 3);
            tdm_load_2d(smem + KM_OFF + (cur ^ 1) * KM_SZ,
                        kmb + jn, 32, 256, HW, 256, HW, 3, 3);
        }

        const bf16* knL = (const bf16*)(smem + KN_OFF + cur * KN_SZ);
        const bf16* kmL = (const bf16*)(smem + KM_OFF + cur * KM_SZ);

        // ---------- GEMM1: two 16x16 S tiles over K=256 ----------
        // A layout (ISA 16-bit A 16x32): lane half -> K {0..7,16..23}/{8..15,24..31}
        // Software-pipelined: chunk k+1's operands are fetched before wmma(k).
        const bf16* aRow0 = knL + m16 * 264 + half * 8;          // 528B rows
        const bf16* aRow1 = aRow0 + 16 * 264;

        bf16x8 a0l = *(const bf16x8*)&aRow0[0];
        bf16x8 a0h = *(const bf16x8*)&aRow0[16];
        bf16x8 a1l = *(const bf16x8*)&aRow1[0];
        bf16x8 a1h = *(const bf16x8*)&aRow1[16];

        f32x8 s0 = (f32x8){0.f,0.f,0.f,0.f,0.f,0.f,0.f,0.f};
        f32x8 s1 = s0;
        #pragma unroll
        for (int k = 0; k < 8; ++k) {
            const bf16x16 A0 = cat16(a0l, a0h);
            const bf16x16 A1 = cat16(a1l, a1h);
            if (k < 7) {
                const int c1 = (k + 1) * 32;
                a0l = *(const bf16x8*)&aRow0[c1];
                a0h = *(const bf16x8*)&aRow0[c1 + 16];
                a1l = *(const bf16x8*)&aRow1[c1];
                a1h = *(const bf16x8*)&aRow1[c1 + 16];
            }
            s0 = __builtin_amdgcn_wmma_f32_16x16x32_bf16(false, A0, false, bq[k],
                                                         (short)0, s0, false, false);
            s1 = __builtin_amdgcn_wmma_f32_16x16x32_bf16(false, A1, false, bq[k],
                                                         (short)0, s1, false, false);
        }

        // ---------- online softmax over the 32 new keys ----------
        float lmax = s0[0];
        #pragma unroll
        for (int r = 0; r < 8; ++r) { lmax = fmaxf(lmax, s0[r]); lmax = fmaxf(lmax, s1[r]); }
        lmax = fmaxf(lmax, __shfl_xor(lmax, 16, 32));   // column i in lanes n, n+16
        const float m_new = fmaxf(m_run, lmax);
        const float scale = __expf(m_run - m_new);

        float p0[8], p1[8];
        float lsum = 0.f;
        #pragma unroll
        for (int r = 0; r < 8; ++r) {
            p0[r] = __expf(s0[r] - m_new);
            p1[r] = __expf(s1[r] - m_new);
            lsum += p0[r] + p1[r];
        }
        lsum += __shfl_xor(lsum, 16, 32);
        l_run = l_run * scale + lsum;
        m_run = m_new;
        #pragma unroll
        for (int tt = 0; tt < 16; ++tt) acc[tt] *= scale;

        // ---------- build P as GEMM2 B operand (K=j packed per lane half) ----
        bf16x16 pB;
        #pragma unroll
        for (int r = 0; r < 8; ++r) {
            const float sx0 = __shfl_xor(p0[r], 16, 32);
            const float sx1 = __shfl_xor(p1[r], 16, 32);
            const float lo  = p0[r];
            const float hi  = p1[r];
            pB[r]     = (bf16)lo;
            pB[r + 8] = (bf16)hi;
        }

        // ---------- GEMM2: accumulate O over 16 channel tiles ----------
        // Software-pipelined: tile ct+1's A operand fetched before wmma(ct).
        const bf16* kRow = kmL + m16 * 40 + half * 8;            // 80B rows
        bf16x8 kl = *(const bf16x8*)&kRow[0];
        bf16x8 kh = *(const bf16x8*)&kRow[16];
        #pragma unroll
        for (int ct = 0; ct < 16; ++ct) {
            const bf16x16 kA = cat16(kl, kh);
            if (ct < 15) {
                const bf16* kp = kRow + (ct + 1) * 16 * 40;
                kl = *(const bf16x8*)&kp[0];
                kh = *(const bf16x8*)&kp[16];
            }
            acc[ct] = __builtin_amdgcn_wmma_f32_16x16x32_bf16(false, kA, false, pB,
                                                              (short)0, acc[ct], false, false);
        }

        // ---- publish next buffer: DMA done (wave0) + all waves done reading ----
        if (wave == 0) __builtin_amdgcn_s_wait_tensorcnt(0);
        __syncthreads();
    }

    // ---------- epilogue: normalize, blend with mask, skip override ----------
    const int   icol = i0 + m16;
    const float mcol = mask[(size_t)b * HW + icol];
    const bool  skip = msum[b] > (float)(HW - 10);
    const float fa   = skip ? 1.f : mcol;    // out = o*(1-fa) + F*fa
    const float invl = 1.f / l_run;

    #pragma unroll
    for (int ct = 0; ct < 16; ++ct) {
        #pragma unroll
        for (int r = 0; r < 8; ++r) {
            const int c = ct * 16 + half * 8 + r;      // D layout: M = r + 8*half
            const size_t idx = ((size_t)b * CH + c) * HW + icol;
            const float fv = F[idx];
            const float o  = acc[ct][r] * invl;
            out[idx] = o * (1.f - fa) + fv * fa;
        }
    }
}

// ---------------------------------------------------------------------------
extern "C" void kernel_launch(void* const* d_in, const int* in_sizes, int n_in,
                              void* d_out, int out_size, void* d_ws, size_t ws_size,
                              hipStream_t stream) {
    (void)in_sizes; (void)n_in; (void)out_size; (void)ws_size;

    const float* F    = (const float*)d_in[0];   // (4,256,64,64) fp32
    const float* mask = (const float*)d_in[1];   // (4,1,64,64)  fp32
    float*       out  = (float*)d_out;           // (4,256,64,64) fp32

    // Workspace carve-up (~24.2 MB total)
    char* ws = (char*)d_ws;
    const size_t bfN = (size_t)BATCH * HW * CH * sizeof(bf16);   // 8 MB each
    bf16*  Kn    = (bf16*)ws;            ws += bfN;
    bf16*  FTb   = (bf16*)ws;            ws += bfN;
    bf16*  KmT   = (bf16*)ws;            ws += bfN;
    float* invn  = (float*)ws;           ws += (size_t)BATCH * HW * sizeof(float);
    float* invnm = (float*)ws;           ws += (size_t)BATCH * HW * sizeof(float);
    float* msum  = (float*)ws;

    prep_masksum<<<BATCH, 256, 0, stream>>>(mask, msum);
    prep_norm<<<dim3(HW / 256, BATCH), 256, 0, stream>>>(F, mask, invn, invnm);
    prep_transpose<<<dim3(HW / 32, CH / 32, BATCH), dim3(32, 8), 0, stream>>>(
        F, invn, invnm, Kn, FTb, KmT);
    attn_kernel<<<dim3(HW / 64, BATCH), 128, SMEM_SZ, stream>>>(
        Kn, FTb, KmT, F, mask, msum, out);
}
